// AttentionCritic_87033217286396
// MI455X (gfx1250) — matrix operand order repaired and checked by reading, not compiled, not run on hardware
//
#include <hip/hip_runtime.h>

// AttentionCritic for MI455X (gfx1250, wave32, WMMA).
// Algebraic restructuring: the reference's [N,N,E] ctx tensor and the
// (N*N)xExE projection collapse through linearity of the masked j-sum,
// reducing ~44 GFLOP to ~1.0 GFLOP of small GEMMs, all L2-resident, all
// running through v_wmma_f32_16x16x32_f16 (fp32 accumulate).
// Register blocking: one wave computes a 16x(16*NTILE) tile (NTILE
// accumulators). This revision hoists all NTILE B-chunk loads ahead of the
// WMMA chain in each k-step so load latency overlaps the preceding WMMAs
// instead of stalling before every one. A is read with aligned b128 vector
// loads; B is pre-packed into the WMMA wave32 lane layout (f16, zero-padded
// K); the K=144 tail is handled branch-free.

#define NAGENT 256
#define HIDDIM 128
#define DDIM   144   // concat dim
#define EDIM   576   // embed dim
#define NHEAD  4
#define HDIM   144   // per-head dim
#define ACTDIM 5
#define OBS_RX 4
#define OBS_RY 2

typedef __attribute__((ext_vector_type(16))) _Float16 v16h;
typedef __attribute__((ext_vector_type(8)))  float    v8f;
typedef __attribute__((ext_vector_type(4)))  float    v4f;

__device__ __forceinline__ float wred_max32(float v) {
#pragma unroll
  for (int o = 16; o > 0; o >>= 1) v = fmaxf(v, __shfl_xor(v, o, 32));
  return v;
}
__device__ __forceinline__ float wred_sum32(float v) {
#pragma unroll
  for (int o = 16; o > 0; o >>= 1) v += __shfl_xor(v, o, 32);
  return v;
}

// -------------------------------------------------------------------------
// Pack B[K,N] (fp32, row-major, optionally batched) into the WMMA wave32
// B-operand layout as f16, zero-filling K up to nks*32:
//   Bp[((tn*nks + ks)*32 + lane)*16 + e]  <-  B[ks*32 + (lane>>4)*16 + e,
//                                               tn*16 + (lane&15)]
// Loads are clamped+selected (no exec-masked loads).
// -------------------------------------------------------------------------
__global__ void pack_b_kernel(const float* __restrict__ B, int ldb, long sB,
                              _Float16* __restrict__ Bp, long sBp,
                              int tilesN, int K, int nks) {
  const int b = blockIdx.y;
  B  += (long)b * sB;
  Bp += (long)b * sBp;
  const int idx = blockIdx.x * blockDim.x + threadIdx.x;
  if (idx >= tilesN * nks * 32) return;
  const int lane  = idx & 31;
  const int rest  = idx >> 5;
  const int ks    = rest % nks;
  const int tn    = rest / nks;
  const int col   = (tn << 4) + (lane & 15);
  const int kbase = (ks << 5) + ((lane >> 4) << 4);
  v16h out;
#pragma unroll
  for (int e = 0; e < 16; ++e) {
    const int kb = kbase + e;
    const int kc = kb < K ? kb : (K - 1);         // clamped, always-safe load
    const float v = B[(long)kc * ldb + col];
    out[e] = (_Float16)(kb < K ? v : 0.0f);       // value select, no exec mask
  }
  *(v16h*)(Bp + (long)idx * 16) = out;
}

// -------------------------------------------------------------------------
// WMMA GEMM. One wave = one 16x(16*NTILE) tile (NTILE f32 accumulators).
//   TRANSB=0 : Out = alpha*(A @ B) + bias(*rowscale) ; B comes pre-packed f16
//   TRANSB=1 : Out = alpha*(A @ B^T) + bias          ; B read directly (f32)
//   TAIL16   : K % 32 == 16 (only K=144 here); tail step is branch-free:
//              A lanes with hl==1 are uniformly OOB -> clamped address * 0; NT-B lanes with
//              hl==1 are uniformly OOB -> clamped address * 0.
// All B chunks for a k-step are loaded before the WMMA chain so load latency
// overlaps WMMA execution. A/B both use contiguous k order (k = 16*half + e).
// -------------------------------------------------------------------------
template <bool TRANSB, bool TAIL16, int NTILE>
__global__ void gemm_wmma_kernel(const float* __restrict__ A, int lda, long sA,
                                 const float* __restrict__ Bf, int ldb, long sB,
                                 const _Float16* __restrict__ Bp, long sBp,
                                 float* __restrict__ Out, int ldo, long sO,
                                 int M, int N, int K, int nks,
                                 const float* __restrict__ bias,
                                 const float* __restrict__ rowscale,
                                 float alpha) {
  const int b = blockIdx.y;
  A   += (long)b * sA;
  if (TRANSB) Bf += (long)b * sB; else Bp += (long)b * sBp;
  Out += (long)b * sO;

  const int wavesPerBlock = blockDim.x >> 5;
  const int wave    = threadIdx.x >> 5;
  const int tilesNg = N / (16 * NTILE);
  const int tiles   = (M >> 4) * tilesNg;
  const int t       = blockIdx.x * wavesPerBlock + wave;
  if (t >= tiles) return;                    // whole-wave uniform
  const int tm  = t / tilesNg;
  const int tn0 = (t % tilesNg) * NTILE;

  const int lane = threadIdx.x & 31;
  const int hl   = lane >> 4;                // lane half
  const int l15  = lane & 15;
  const int rowA = (tm << 4) + l15;

  const int Kfloor = TAIL16 ? (K - 16) : K;
  // A row base incl. lane-half K offset; all strides keep 16B alignment.
  const float* arow = A + (long)rowA * lda + (hl << 4);
  // Packed-B chunk base for subtile 0 (chunks are 512 halves each).
  const _Float16* bpt = TRANSB ? nullptr
                               : Bp + (long)(tn0 * nks) * 512 + (long)lane * 16;

  v8f acc[NTILE];
#pragma unroll
  for (int c = 0; c < NTILE; ++c) acc[c] = (v8f){};

  int ks = 0;
  for (int k0 = 0; k0 < Kfloor; k0 += 32, ++ks) {
    // ---- A 16x32: elems 0..15 <- K hl*16+0..15
    const v4f a0 = *(const v4f*)(arow + k0);
    const v4f a1 = *(const v4f*)(arow + k0 + 4);
    const v4f a2 = *(const v4f*)(arow + k0 + 8);
    const v4f a3 = *(const v4f*)(arow + k0 + 12);
    v16h av;
#pragma unroll
    for (int e = 0; e < 4; ++e) {
      av[e]      = (_Float16)a0[e];
      av[e + 4]  = (_Float16)a1[e];
      av[e + 8]  = (_Float16)a2[e];
      av[e + 12] = (_Float16)a3[e];
    }
    // ---- stage ALL B chunks for this k-step, then run the WMMA chain.
    v16h bvs[NTILE];
#pragma unroll
    for (int c = 0; c < NTILE; ++c) {
      if (TRANSB) {
        const int colB = ((tn0 + c) << 4) + l15;
        const float* brow = Bf + (long)colB * ldb + (hl << 4) + k0;
        const v4f b0 = *(const v4f*)(brow);
        const v4f b1 = *(const v4f*)(brow + 4);
        const v4f b2 = *(const v4f*)(brow + 8);
        const v4f b3 = *(const v4f*)(brow + 12);
#pragma unroll
        for (int e = 0; e < 4; ++e) {
          bvs[c][e]      = (_Float16)b0[e];
          bvs[c][e + 4]  = (_Float16)b1[e];
          bvs[c][e + 8]  = (_Float16)b2[e];
          bvs[c][e + 12] = (_Float16)b3[e];
        }
      } else {
        bvs[c] = *(const v16h*)(bpt + (long)(c * nks + ks) * 512);
      }
    }
#pragma unroll
    for (int c = 0; c < NTILE; ++c)
      acc[c] = __builtin_amdgcn_wmma_f32_16x16x32_f16(
          false, av, false, bvs[c], (short)0, acc[c], false, false);
  }

  if (TAIL16) {
    // A: hl==0 lanes K = Kfloor..Kfloor+15 valid; hl==1 lanes uniformly OOB -> clamped * 0.
    const float za = hl ? 0.0f : 1.0f;
    const float* at = arow + Kfloor - (hl << 4);
    const v4f a0 = *(const v4f*)(at);
    const v4f a1 = *(const v4f*)(at + 4);
    const v4f a2 = *(const v4f*)(at + 8);
    const v4f a3 = *(const v4f*)(at + 12);
    v16h av;
#pragma unroll
    for (int e = 0; e < 4; ++e) {
      av[e]      = (_Float16)(a0[e] * za);
      av[e + 4]  = (_Float16)(a1[e] * za);
      av[e + 8]  = (_Float16)(a2[e] * za);
      av[e + 12] = (_Float16)(a3[e] * za);
    }
    v16h bvs[NTILE];
#pragma unroll
    for (int c = 0; c < NTILE; ++c) {
      if (TRANSB) {
        // hl==0 lanes: K = Kfloor..Kfloor+15 valid. hl==1 lanes: uniformly
        // OOB -> read a clamped in-range address, multiply by 0 (branch-free).
        const int colB = ((tn0 + c) << 4) + l15;
        const float zm = hl ? 0.0f : 1.0f;
        const float* brow = Bf + (long)colB * ldb + (hl ? Kfloor - 16 : Kfloor);
        const v4f b0 = *(const v4f*)(brow);
        const v4f b1 = *(const v4f*)(brow + 4);
        const v4f b2 = *(const v4f*)(brow + 8);
        const v4f b3 = *(const v4f*)(brow + 12);
#pragma unroll
        for (int e = 0; e < 4; ++e) {
          bvs[c][e]      = (_Float16)(b0[e] * zm);
          bvs[c][e + 4]  = (_Float16)(b1[e] * zm);
          bvs[c][e + 8]  = (_Float16)(b2[e] * zm);
          bvs[c][e + 12] = (_Float16)(b3[e] * zm);
        }
      } else {
        bvs[c] = *(const v16h*)(bpt + (long)(c * nks + ks) * 512);  // 0-padded
      }
    }
#pragma unroll
    for (int c = 0; c < NTILE; ++c)
      acc[c] = __builtin_amdgcn_wmma_f32_16x16x32_f16(
          false, av, false, bvs[c], (short)0, acc[c], false, false);
  }

#pragma unroll
  for (int c = 0; c < NTILE; ++c) {
    const int colB = ((tn0 + c) << 4) + l15;
    const float bterm = bias ? bias[colB] : 0.0f;
#pragma unroll
    for (int r = 0; r < 8; ++r) {
      const int m = (tm << 4) + r + (hl << 3);   // C/D layout: M = r + 8*half
      float bt = bterm;
      if (rowscale) bt *= rowscale[m];
      Out[(long)m * ldo + colB] = alpha * acc[c][r] + bt;
    }
  }
}

// -------------------------------------------------------------------------
// mask[i,j] = (j>i) && |xi-xj|<=4 && |yi-yj|<=2 ;  cnt[i] = popcount(row i)
// -------------------------------------------------------------------------
__global__ void mask_kernel(const int* __restrict__ state,
                            unsigned char* __restrict__ mask,
                            float* __restrict__ cnt) {
  const int i = blockIdx.x;
  const int j = threadIdx.x;
  const int xi = state[2 * i], yi = state[2 * i + 1];
  const int xj = state[2 * j], yj = state[2 * j + 1];
  int dx = xi - xj; dx = dx < 0 ? -dx : dx;
  int dy = yi - yj; dy = dy < 0 ? -dy : dy;
  const int m = (j > i) && (dx <= OBS_RX) && (dy <= OBS_RY);
  mask[i * NAGENT + j] = (unsigned char)m;

  __shared__ int red[NAGENT];
  red[j] = m;
  __syncthreads();
#pragma unroll
  for (int s = NAGENT / 2; s > 0; s >>= 1) {
    if (j < s) red[j] += red[j + s];
    __syncthreads();
  }
  if (j == 0) cnt[i] = (float)red[0];
}

// -------------------------------------------------------------------------
__global__ void concat_action_kernel(const float* __restrict__ action,
                                     float* __restrict__ C) {
  const int idx = blockIdx.x * blockDim.x + threadIdx.x;  // N*HIDDIM
  const int i = idx >> 7;
  const int j = idx & 127;
  C[i * DDIM + 16 + j] = action[idx];
}

// -------------------------------------------------------------------------
// w[h,i,k] = sum_{j in O_i} softmax_{k in O_i}(S[h,j,:])[k]
// One block per agent i; one wave per head; lane owns 8 k-slots.
// Mask sparsity (avg |O_i| ~ 4) makes this cheap; wave32 shuffle reductions.
// -------------------------------------------------------------------------
__global__ void attn_w_kernel(const float* __restrict__ S,
                              const unsigned char* __restrict__ mask,
                              float* __restrict__ W) {
  const int i   = blockIdx.x;
  const int tid = threadIdx.x;           // 128 = 4 waves = 4 heads
  __shared__ unsigned char mrow[NAGENT];
  if (tid < NAGENT / 4)
    ((unsigned int*)mrow)[tid] = ((const unsigned int*)(mask + i * NAGENT))[tid];
  __syncthreads();

  const int h    = tid >> 5;
  const int lane = tid & 31;
  const float* Sh = S + (long)h * NAGENT * NAGENT;

  float mk[8], wacc[8];
#pragma unroll
  for (int r = 0; r < 8; ++r) {
    mk[r]   = mrow[lane + 32 * r] ? 1.0f : 0.0f;
    wacc[r] = 0.0f;
  }

  for (int j = 0; j < NAGENT; ++j) {
    if (!mrow[j]) continue;              // uniform across the wave
    const float* row = Sh + (long)j * NAGENT;
    float s[8];
    float lm = -1e30f;
#pragma unroll
    for (int r = 0; r < 8; ++r) {
      s[r] = row[lane + 32 * r];
      lm   = fmaxf(lm, mk[r] ? s[r] : -1e30f);
    }
    const float m = wred_max32(lm);
    float ls = 0.0f, p[8];
#pragma unroll
    for (int r = 0; r < 8; ++r) {
      p[r] = mk[r] * __expf(s[r] - m);
      ls  += p[r];
    }
    const float inv = 1.0f / fmaxf(wred_sum32(ls), 1e-9f);
#pragma unroll
    for (int r = 0; r < 8; ++r) wacc[r] += p[r] * inv;
  }

  float* Wo = W + (long)h * NAGENT * NAGENT + (long)i * NAGENT;
#pragma unroll
  for (int r = 0; r < 8; ++r) Wo[lane + 32 * r] = wacc[r];
}

// -------------------------------------------------------------------------
__global__ void dueling_kernel(const float* __restrict__ H,
                               const float* __restrict__ W_val,
                               const float* __restrict__ b_val,
                               const float* __restrict__ W_adv,
                               const float* __restrict__ b_adv,
                               float* __restrict__ out) {
  const int i = threadIdx.x;
  const float* h = H + i * DDIM;
  float v = b_val[0];
  float a[ACTDIM];
#pragma unroll
  for (int x = 0; x < ACTDIM; ++x) a[x] = b_adv[x];
  for (int d = 0; d < DDIM; ++d) {
    const float hv = h[d];
    v += hv * W_val[d];
#pragma unroll
    for (int x = 0; x < ACTDIM; ++x) a[x] += hv * W_adv[d * ACTDIM + x];
  }
  float mean = 0.0f;
#pragma unroll
  for (int x = 0; x < ACTDIM; ++x) mean += a[x];
  mean *= (1.0f / ACTDIM);
#pragma unroll
  for (int x = 0; x < ACTDIM; ++x) out[i * ACTDIM + x] = v + a[x] - mean;
}

// -------------------------------------------------------------------------
extern "C" void kernel_launch(void* const* d_in, const int* in_sizes, int n_in,
                              void* d_out, int out_size, void* d_ws, size_t ws_size,
                              hipStream_t stream) {
  (void)in_sizes; (void)n_in; (void)out_size; (void)ws_size;
  const float* hidden  = (const float*)d_in[0];
  const float* action  = (const float*)d_in[1];
  const int*   state   = (const int*)  d_in[2];
  const float* W_enc   = (const float*)d_in[3];
  const float* b_enc   = (const float*)d_in[4];
  const float* Wq      = (const float*)d_in[5];
  const float* bq      = (const float*)d_in[6];
  const float* Wk      = (const float*)d_in[7];
  const float* bk      = (const float*)d_in[8];
  const float* Wv      = (const float*)d_in[9];
  const float* bvv     = (const float*)d_in[10];
  const float* Wiq     = (const float*)d_in[11];
  const float* biq     = (const float*)d_in[12];
  const float* Wik     = (const float*)d_in[13];
  const float* bik     = (const float*)d_in[14];
  const float* Wiv     = (const float*)d_in[15];
  const float* biv     = (const float*)d_in[16];
  const float* Wo_proj = (const float*)d_in[17];
  const float* bo_proj = (const float*)d_in[18];
  const float* W_O     = (const float*)d_in[19];
  const float* W_val   = (const float*)d_in[20];
  const float* b_val   = (const float*)d_in[21];
  const float* W_adv   = (const float*)d_in[22];
  const float* b_adv   = (const float*)d_in[23];
  float* out = (float*)d_out;

  // Workspace layout (floats); ~6.7 MB total, fully overwritten every call.
  float* ws  = (float*)d_ws;
  float* C   = ws;              // [256,144]
  float* T   = C   + 36864;     // [256,576] scratch
  float* Qb  = T   + 147456;    // [256,576]
  float* Kb  = Qb  + 147456;    // [256,576]
  float* Vb  = Kb  + 147456;    // [256,576]
  float* S   = Vb  + 147456;    // [4,256,256]
  float* Wt  = S   + 262144;    // [4,256,256]
  float* HC  = Wt  + 262144;    // [256,576] hctx
  float* T2  = HC  + 147456;    // [256,576]
  float* Hb  = T2  + 147456;    // [256,144]
  float* CNT = Hb  + 36864;     // [256]
  _Float16* BPH = (_Float16*)(CNT + 256);             // packed B, <=331776 h
  unsigned char* MASK = (unsigned char*)(CNT + 256 + 165888); // [256,256] u8

  auto packB = [&](const float* B, int ldb, long sB, long sBp,
                   int N, int K, int nks, int batch) {
    const int total = (N / 16) * nks * 32;
    dim3 grid((total + 255) / 256, batch);
    pack_b_kernel<<<grid, 256, 0, stream>>>(B, ldb, sB, BPH, sBp, N / 16, K, nks);
  };

  // NN GEMM with packed B (TAIL16 by K, NTILE by N)
  auto gemmNN = [&](const float* A, int lda, long sA, long sBp,
                    float* O, int ldo, long sO,
                    int M, int N, int K, int nks,
                    const float* bias, const float* rowscale, int batch) {
    const int nt    = (N % 64 == 0) ? 4 : ((N % 48 == 0) ? 3 : 1);
    const int tiles = (M / 16) * (N / (16 * nt));
    dim3 grid((tiles + 3) / 4, batch);
    dim3 block(128);  // 4 waves
    const bool tail = (K & 31) != 0;
    if (nt == 4) {
      if (tail)
        gemm_wmma_kernel<false, true, 4><<<grid, block, 0, stream>>>(
            A, lda, sA, nullptr, 0, 0, BPH, sBp, O, ldo, sO,
            M, N, K, nks, bias, rowscale, 1.0f);
      else
        gemm_wmma_kernel<false, false, 4><<<grid, block, 0, stream>>>(
            A, lda, sA, nullptr, 0, 0, BPH, sBp, O, ldo, sO,
            M, N, K, nks, bias, rowscale, 1.0f);
    } else if (nt == 3) {
      if (tail)
        gemm_wmma_kernel<false, true, 3><<<grid, block, 0, stream>>>(
            A, lda, sA, nullptr, 0, 0, BPH, sBp, O, ldo, sO,
            M, N, K, nks, bias, rowscale, 1.0f);
      else
        gemm_wmma_kernel<false, false, 3><<<grid, block, 0, stream>>>(
            A, lda, sA, nullptr, 0, 0, BPH, sBp, O, ldo, sO,
            M, N, K, nks, bias, rowscale, 1.0f);
    } else {
      if (tail)
        gemm_wmma_kernel<false, true, 1><<<grid, block, 0, stream>>>(
            A, lda, sA, nullptr, 0, 0, BPH, sBp, O, ldo, sO,
            M, N, K, nks, bias, rowscale, 1.0f);
      else
        gemm_wmma_kernel<false, false, 1><<<grid, block, 0, stream>>>(
            A, lda, sA, nullptr, 0, 0, BPH, sBp, O, ldo, sO,
            M, N, K, nks, bias, rowscale, 1.0f);
    }
  };

  // 1) FOV mask + neighbor counts
  mask_kernel<<<NAGENT, NAGENT, 0, stream>>>(state, MASK, CNT);

  // 2) C = concat(hidden @ W_enc + b_enc, action)     [256,144]
  packB(W_enc, 16, 0, 0, 16, HIDDIM, 4, 1);
  gemmNN(hidden, HIDDIM, 0, 0, C, DDIM, 0, NAGENT, 16, HIDDIM, 4, b_enc, nullptr, 1);
  concat_action_kernel<<<(NAGENT * HIDDIM) / 256, 256, 0, stream>>>(action, C);

  // 3) q/k/v = (C@W + b) @ Wi + bi                    [256,576] each
  packB(Wq, EDIM, 0, 0, EDIM, DDIM, 5, 1);
  gemmNN(C, DDIM, 0, 0, T, EDIM, 0, NAGENT, EDIM, DDIM, 5, bq, nullptr, 1);
  packB(Wiq, EDIM, 0, 0, EDIM, EDIM, 18, 1);
  gemmNN(T, EDIM, 0, 0, Qb, EDIM, 0, NAGENT, EDIM, EDIM, 18, biq, nullptr, 1);

  packB(Wk, EDIM, 0, 0, EDIM, DDIM, 5, 1);
  gemmNN(C, DDIM, 0, 0, T, EDIM, 0, NAGENT, EDIM, DDIM, 5, bk, nullptr, 1);
  packB(Wik, EDIM, 0, 0, EDIM, EDIM, 18, 1);
  gemmNN(T, EDIM, 0, 0, Kb, EDIM, 0, NAGENT, EDIM, EDIM, 18, bik, nullptr, 1);

  packB(Wv, EDIM, 0, 0, EDIM, DDIM, 5, 1);
  gemmNN(C, DDIM, 0, 0, T, EDIM, 0, NAGENT, EDIM, DDIM, 5, bvv, nullptr, 1);
  packB(Wiv, EDIM, 0, 0, EDIM, EDIM, 18, 1);
  gemmNN(T, EDIM, 0, 0, Vb, EDIM, 0, NAGENT, EDIM, EDIM, 18, biv, nullptr, 1);

  // 4) S[h] = q_h @ k_h^T / sqrt(144)   (NT GEMM, batched over 4 heads)
  {
    const int tiles = (NAGENT / 16) * (NAGENT / 64);   // NTILE=4
    dim3 grid((tiles + 3) / 4, NHEAD);
    gemm_wmma_kernel<true, true, 4><<<grid, 128, 0, stream>>>(
        Qb, EDIM, HDIM, Kb, EDIM, HDIM, nullptr, 0,
        S, NAGENT, (long)NAGENT * NAGENT,
        NAGENT, NAGENT, HDIM, 5, nullptr, nullptr, 1.0f / 12.0f);
  }

  // 5) w[h,i,k] = masked softmax aggregate
  attn_w_kernel<<<NAGENT, 128, 0, stream>>>(S, MASK, Wt);

  // 6) hctx_h = w_h @ v_h               (batched over heads, packed v)
  packB(Vb, EDIM, HDIM, 36864, HDIM, NAGENT, 8, NHEAD);
  gemmNN(Wt, NAGENT, (long)NAGENT * NAGENT, 36864, HC, EDIM, HDIM,
         NAGENT, HDIM, NAGENT, 8, nullptr, nullptr, NHEAD);

  // 7) h = (hctx @ Wo_proj + cnt_i * bo_proj) @ W_O
  packB(Wo_proj, EDIM, 0, 0, EDIM, EDIM, 18, 1);
  gemmNN(HC, EDIM, 0, 0, T2, EDIM, 0, NAGENT, EDIM, EDIM, 18, bo_proj, CNT, 1);
  packB(W_O, DDIM, 0, 0, DDIM, EDIM, 18, 1);
  gemmNN(T2, EDIM, 0, 0, Hb, DDIM, 0, NAGENT, DDIM, EDIM, 18, nullptr, nullptr, 1);

  // 8) dueling head -> Q [256,5]
  dueling_kernel<<<1, NAGENT, 0, stream>>>(Hb, W_val, b_val, W_adv, b_adv, out);
}
